// _RVarDSepBlock_80229989089634
// MI455X (gfx1250) — matrix operand order repaired and checked by reading, not compiled, not run on hardware
//
#include <hip/hip_runtime.h>

// ---------------------------------------------------------------------------
// Types for CDNA5 WMMA
// ---------------------------------------------------------------------------
typedef __attribute__((ext_vector_type(16))) __bf16  v16bf;
typedef __attribute__((ext_vector_type(8)))  float   v8f;
typedef __attribute__((ext_vector_type(4)))  unsigned int u32x4;
typedef __attribute__((ext_vector_type(4)))  float   f32x4;

union ABf16 { v16bf v; u32x4 q[2]; };
union Acc8  { v8f   v; u32x4 q[2]; };
union HPack { __bf16 h[2]; unsigned int u; };

#define T_LEN   2048
#define B_SZ    64
#define IN_CH   64
#define OUT_CH  128
#define GATES   512            // 4*OUT_CH

// workspace layout (bytes)
#define GX_BYTES   ((size_t)T_LEN * B_SZ * GATES * 4)        // 256 MB, f32, C-operand swizzled
#define H1T_BYTES  ((size_t)T_LEN * B_SZ * IN_CH * 2)        // 16 MB,  bf16 [T][B][64]
#define MW_BYTES   ((size_t)32 * 2 * 32 * 16 * 2)            // 64 KB swizzled M
#define WHH_BYTES  ((size_t)32 * 4 * 32 * 16 * 2)            // 128 KB swizzled w_hh
#define BETA_BYTES ((size_t)GATES * 4)

__device__ __forceinline__ float sigf(float x) {
    return 1.0f / (1.0f + __expf(-x));
}
__device__ __forceinline__ float tanhfast(float x) {
    float e = __expf(2.0f * x);
    return 1.0f - 2.0f / (e + 1.0f);       // safe at +/-inf
}

// ---------------------------------------------------------------------------
// Kernel A: depthwise conv(K=3,pad=1) + BN(inference) + ReLU + residual.
// Writes h1 transposed to [T][B][64] bf16 (GEMM-row-major, K=c contiguous).
// grid: B*32 blocks (b, 64-wide t-chunk), 256 threads.
// ---------------------------------------------------------------------------
__global__ __launch_bounds__(256) void front_kernel(
    const float* __restrict__ x, const float* __restrict__ dw_w,
    const float* __restrict__ dw_b, const float* __restrict__ bn_gamma,
    const float* __restrict__ bn_beta, const float* __restrict__ bn_mean,
    const float* __restrict__ bn_var, __bf16* __restrict__ h1t)
{
    __shared__ float xs[IN_CH][66];     // cols: t0-1 .. t0+64
    const int b  = blockIdx.x >> 5;
    const int t0 = (blockIdx.x & 31) * 64;
    const int tid = threadIdx.x;

    for (int idx = tid; idx < IN_CH * 66; idx += 256) {
        int c = idx / 66, tt = idx % 66;
        int t = t0 + tt - 1;
        xs[c][tt] = (t >= 0 && t < T_LEN) ? x[((size_t)(b * IN_CH + c)) * T_LEN + t] : 0.0f;
    }
    __syncthreads();

    const int c = tid & 63;
    const int q = tid >> 6;             // 0..3, each covers 16 t's
    const float w0 = dw_w[c * 3 + 0], w1 = dw_w[c * 3 + 1], w2 = dw_w[c * 3 + 2];
    const float bias = dw_b[c];
    const float inv  = bn_gamma[c] * rsqrtf(bn_var[c] + 1e-5f);
    const float sh   = bn_beta[c] - bn_mean[c] * inv;

    #pragma unroll
    for (int i = 0; i < 16; ++i) {
        int ti = q * 16 + i;
        float conv = w0 * xs[c][ti] + w1 * xs[c][ti + 1] + w2 * xs[c][ti + 2] + bias;
        float h = conv * inv + sh;
        h = fmaxf(h, 0.0f) + xs[c][ti + 1];
        h1t[((size_t)(t0 + ti) * B_SZ + b) * IN_CH + c] = (__bf16)h;
    }
}

// ---------------------------------------------------------------------------
// Prep 1: beta[g] = b_ih[g] + b_hh[g] + w_ih[g,:] . pw_b
// ---------------------------------------------------------------------------
__global__ __launch_bounds__(512) void beta_kernel(
    const float* __restrict__ w_ih, const float* __restrict__ pw_b,
    const float* __restrict__ b_ih, const float* __restrict__ b_hh,
    float* __restrict__ beta)
{
    int g = threadIdx.x;
    float s = b_ih[g] + b_hh[g];
    for (int o = 0; o < OUT_CH; ++o) s += w_ih[g * OUT_CH + o] * pw_b[o];
    beta[g] = s;
}

// ---------------------------------------------------------------------------
// Prep 2: M = w_ih @ pw_w  ([512,64]) in WMMA-B swizzled bf16 order:
//   flat idx = ((nt*2+kk)*32 + lane)*16 + s,
//   value    = M[nt*16 + lane%16][kk*32 + 16*(lane/16) + s]
// ---------------------------------------------------------------------------
__global__ __launch_bounds__(1024) void mw_kernel(
    const float* __restrict__ w_ih, const float* __restrict__ pw_w,
    __bf16* __restrict__ MwOp)
{
    int idx = blockIdx.x * 1024 + threadIdx.x;     // 32768 values
    int s    = idx & 15;
    int lane = (idx >> 4) & 31;
    int kk   = (idx >> 9) & 1;
    int nt   = idx >> 10;
    int g = nt * 16 + (lane & 15);
    int c = kk * 32 + 16 * (lane >> 4) + s;
    float v = 0.0f;
    for (int o = 0; o < OUT_CH; ++o) v += w_ih[g * OUT_CH + o] * pw_w[o * IN_CH + c];
    MwOp[idx] = (__bf16)v;
}

// ---------------------------------------------------------------------------
// Prep 3: w_hh into WMMA-B swizzled bf16 order (4 k-chunks of 32 over K=128):
//   flat idx = ((nt*4+kk)*32 + lane)*16 + s,
//   value    = w_hh[nt*16 + lane%16][kk*32 + 16*(lane/16) + s]
// ---------------------------------------------------------------------------
__global__ __launch_bounds__(1024) void whh_kernel(
    const float* __restrict__ w_hh, __bf16* __restrict__ whhOp)
{
    int idx = blockIdx.x * 1024 + threadIdx.x;     // 65536 values
    int s    = idx & 15;
    int lane = (idx >> 4) & 31;
    int kk   = (idx >> 9) & 3;
    int nt   = idx >> 11;
    int g = nt * 16 + (lane & 15);
    int j = kk * 32 + 16 * (lane >> 4) + s;
    whhOp[idx] = (__bf16)w_hh[g * OUT_CH + j];
}

// ---------------------------------------------------------------------------
// Kernel C: gx GEMM.  gx_tile = h1t_tile @ M^T + beta
//   rows m = t*64+b (131072), N=512, K=64 -> 2x wmma bf16 per 16x16 tile.
// Output stored in WMMA C-operand swizzled order so the LSTM seeds its
// accumulator with two b128 loads:
//   gx_sw[ ((rt*32 + nt)*32 + lane)*8 + v ],  rt = t*4 + b/16
// grid: 8192 row-tiles * 4 n-groups = 32768 blocks, 256 thr = 8 waves.
// ---------------------------------------------------------------------------
__global__ __launch_bounds__(256) void gx_gemm_kernel(
    const __bf16* __restrict__ h1t, const __bf16* __restrict__ MwOp,
    const float* __restrict__ beta, float* __restrict__ gx)
{
    const int rt   = blockIdx.x >> 2;
    const int ngrp = blockIdx.x & 3;
    const int wv   = threadIdx.x >> 5;
    const int lane = threadIdx.x & 31;
    const int nt   = ngrp * 8 + wv;
    const int half = lane >> 4;
    const int n    = lane & 15;
    const size_t m0 = (size_t)rt * 16;

    // A operands: row = m0 + n, K-chunks of 32 -> two 16B loads each
    ABf16 A[2];
    const unsigned char* arow = (const unsigned char*)h1t + (m0 + n) * IN_CH * 2;
    #pragma unroll
    for (int kk = 0; kk < 2; ++kk) {
        A[kk].q[0] = *(const u32x4*)(arow + (kk * 32 + 16 * half) * 2);
        A[kk].q[1] = *(const u32x4*)(arow + (kk * 32 + 16 * half + 8) * 2);
    }
    // B operands (swizzled)
    ABf16 Bop[2];
    const u32x4* bp = (const u32x4*)MwOp;
    #pragma unroll
    for (int kk = 0; kk < 2; ++kk) {
        size_t base = ((size_t)(nt * 2 + kk) * 32 + lane) * 2;
        Bop[kk].q[0] = bp[base + 0];
        Bop[kk].q[1] = bp[base + 1];
    }
    // accumulator = beta broadcast
    float bv = beta[nt * 16 + n];
    Acc8 acc;
    #pragma unroll
    for (int v = 0; v < 8; ++v) acc.v[v] = bv;

    #pragma unroll
    for (int kk = 0; kk < 2; ++kk)
        acc.v = __builtin_amdgcn_wmma_f32_16x16x32_bf16(
            false, A[kk].v, false, Bop[kk].v, (short)0, acc.v, false, false);

    // swizzled store: two b128 per lane
    u32x4* gp = (u32x4*)gx + ((size_t)(rt * 32 + nt) * 32 + lane) * 2;
    gp[0] = acc.q[0];
    gp[1] = acc.q[1];
}

// ---------------------------------------------------------------------------
// Kernel D: persistent LSTM. 4 blocks x 1024 threads; block owns 16 batch rows
// (independent recurrences -> no inter-block sync). Wave wv owns gate tile
// [16b x 16g]; w_hh B-operand kept in registers for all 2048 steps. gx seeds
// are prefetched one step ahead; h outputs buffered 8 steps and flushed as
// b128 stores.
// ---------------------------------------------------------------------------
#define HROW_B 272                      // padded bf16 row stride (bytes)
#define GSTR   516                      // padded gates row stride (floats)

__global__ __launch_bounds__(1024) void lstm_kernel(
    const float* __restrict__ gx, const __bf16* __restrict__ whhOp,
    float* __restrict__ out)
{
    __shared__ __attribute__((aligned(16))) unsigned char hbuf[16 * HROW_B]; // h_prev bf16 [16][128]
    __shared__ float gates[16 * GSTR];                                        // [16][512] padded

    const int tid  = threadIdx.x;
    const int bq   = blockIdx.x;        // batch quarter
    const int b0   = bq * 16;
    const int wv   = tid >> 5;
    const int lane = tid & 31;
    const int half = lane >> 4;
    const int n    = lane & 15;

    // zero h_prev
    for (int i = tid; i < 16 * HROW_B / 4; i += 1024) ((unsigned int*)hbuf)[i] = 0u;

    // resident w_hh B operand: 4 chunks x v16bf = 32 VGPRs
    ABf16 Bop[4];
    {
        const u32x4* bp = (const u32x4*)whhOp;
        #pragma unroll
        for (int kk = 0; kk < 4; ++kk) {
            size_t base = ((size_t)(wv * 4 + kk) * 32 + lane) * 2;
            Bop[kk].q[0] = bp[base + 0];
            Bop[kk].q[1] = bp[base + 1];
        }
    }

    // gx seed pointer for this wave (C-operand swizzled): tile rt = t*4 + bq
    const u32x4* gxp = (const u32x4*)gx + ((size_t)(bq * 32 + wv) * 32 + lane) * 2;
    const size_t gxStep = (size_t)4 * 32 * 32 * 2;   // one t -> 4 row-tiles of 32 nt of 32 lanes of 2 q

    // per-thread cell state: b = tid>>6, hidden units ej, ej+1
    const int eb = tid >> 6;
    const int ej = (tid & 63) * 2;
    float c0 = 0.0f, c1 = 0.0f;
    float h0reg[8], h1reg[8];

    // prefetch t = 0 seed
    u32x4 gq0 = gxp[0];
    u32x4 gq1 = gxp[1];

    __syncthreads();

    for (int t8 = 0; t8 < T_LEN / 8; ++t8) {
        #pragma unroll
        for (int tt = 0; tt < 8; ++tt) {
            const int t = t8 * 8 + tt;

            // ---- phase 1: gates tile = gx_tile + h_prev @ w_hh^T (WMMA) ----
            ABf16 A[4];
            const unsigned char* arow = hbuf + (size_t)n * HROW_B;
            #pragma unroll
            for (int kk = 0; kk < 4; ++kk) {
                A[kk].q[0] = *(const u32x4*)(arow + (kk * 32 + 16 * half) * 2);
                A[kk].q[1] = *(const u32x4*)(arow + (kk * 32 + 16 * half + 8) * 2);
            }
            Acc8 acc;
            acc.q[0] = gq0;
            acc.q[1] = gq1;

            #pragma unroll
            for (int kk = 0; kk < 4; ++kk)
                acc.v = __builtin_amdgcn_wmma_f32_16x16x32_bf16(
                    false, A[kk].v, false, Bop[kk].v, (short)0, acc.v, false, false);

            #pragma unroll
            for (int v = 0; v < 8; ++v)
                gates[(v + 8 * half) * GSTR + wv * 16 + n] = acc.v[v];

            // prefetch next step's gx seed (hides HBM latency behind barrier+gates)
            if (t + 1 < T_LEN) {
                const u32x4* gnext = gxp + (size_t)(t + 1) * gxStep;
                gq0 = gnext[0];
                gq1 = gnext[1];
            }

            __syncthreads();

            // ---- phase 2: elementwise gate math, update c/h ----
            {
                const float* gr = gates + eb * GSTR;
                float i0 = sigf(gr[ej]);
                float f0 = sigf(gr[128 + ej]);
                float g0 = tanhfast(gr[256 + ej]);
                float o0 = sigf(gr[384 + ej]);
                c0 = f0 * c0 + i0 * g0;
                float h0 = o0 * tanhfast(c0);

                float i1 = sigf(gr[ej + 1]);
                float f1 = sigf(gr[128 + ej + 1]);
                float g1 = tanhfast(gr[256 + ej + 1]);
                float o1 = sigf(gr[384 + ej + 1]);
                c1 = f1 * c1 + i1 * g1;
                float h1 = o1 * tanhfast(c1);

                HPack pk; pk.h[0] = (__bf16)h0; pk.h[1] = (__bf16)h1;
                *(unsigned int*)(hbuf + eb * HROW_B + ej * 2) = pk.u;

                h0reg[tt] = h0;
                h1reg[tt] = h1;
            }
            __syncthreads();
        }

        // flush 8 timesteps of h as vector stores (contiguous along T)
        {
            size_t ob = ((size_t)(b0 + eb) * OUT_CH + ej) * T_LEN + (size_t)t8 * 8;
            f32x4 a, b;
            a.x = h0reg[0]; a.y = h0reg[1]; a.z = h0reg[2]; a.w = h0reg[3];
            b.x = h0reg[4]; b.y = h0reg[5]; b.z = h0reg[6]; b.w = h0reg[7];
            *(f32x4*)(out + ob)     = a;
            *(f32x4*)(out + ob + 4) = b;
            a.x = h1reg[0]; a.y = h1reg[1]; a.z = h1reg[2]; a.w = h1reg[3];
            b.x = h1reg[4]; b.y = h1reg[5]; b.z = h1reg[6]; b.w = h1reg[7];
            *(f32x4*)(out + ob + T_LEN)     = a;
            *(f32x4*)(out + ob + T_LEN + 4) = b;
        }
    }
}

// ---------------------------------------------------------------------------
// Host launcher
// ---------------------------------------------------------------------------
extern "C" void kernel_launch(void* const* d_in, const int* in_sizes, int n_in,
                              void* d_out, int out_size, void* d_ws, size_t ws_size,
                              hipStream_t stream) {
    const float* x        = (const float*)d_in[0];
    const float* dw_w     = (const float*)d_in[1];
    const float* dw_b     = (const float*)d_in[2];
    const float* bn_gamma = (const float*)d_in[3];
    const float* bn_beta  = (const float*)d_in[4];
    const float* bn_mean  = (const float*)d_in[5];
    const float* bn_var   = (const float*)d_in[6];
    const float* pw_w     = (const float*)d_in[7];
    const float* pw_b     = (const float*)d_in[8];
    const float* w_ih     = (const float*)d_in[9];
    const float* w_hh     = (const float*)d_in[10];
    const float* b_ih     = (const float*)d_in[11];
    const float* b_hh     = (const float*)d_in[12];
    float* out = (float*)d_out;

    char* ws = (char*)d_ws;
    float*  gx    = (float*)ws;
    __bf16* h1t   = (__bf16*)(ws + GX_BYTES);
    __bf16* MwOp  = (__bf16*)(ws + GX_BYTES + H1T_BYTES);
    __bf16* whhOp = (__bf16*)(ws + GX_BYTES + H1T_BYTES + MW_BYTES);
    float*  beta  = (float*)(ws + GX_BYTES + H1T_BYTES + MW_BYTES + WHH_BYTES);

    front_kernel<<<B_SZ * 32, 256, 0, stream>>>(x, dw_w, dw_b, bn_gamma, bn_beta,
                                                bn_mean, bn_var, h1t);
    beta_kernel<<<1, 512, 0, stream>>>(w_ih, pw_b, b_ih, b_hh, beta);
    mw_kernel<<<32, 1024, 0, stream>>>(w_ih, pw_w, MwOp);
    whh_kernel<<<64, 1024, 0, stream>>>(w_hh, whhOp);
    gx_gemm_kernel<<<(T_LEN * B_SZ / 16) * 4, 256, 0, stream>>>(h1t, MwOp, beta, gx);
    lstm_kernel<<<4, 1024, 0, stream>>>(gx, whhOp, out);
}
